// ASAP_58033598104028
// MI455X (gfx1250) — matrix operand order repaired and checked by reading, not compiled, not run on hardware
//
#include <hip/hip_runtime.h>
#include <hip/hip_bf16.h>

// ============================================================================
// ASAP-pool GNN forward for MI455X (gfx1250, wave32).
// All GEMM-shaped einsums go through v_wmma_f32_16x16x32_bf16 (bf16 in, f32
// accum): the whole working set fits in the 192MB L2, so the model is matrix-
// pipe / latency bound rather than HBM bound, and bf16 WMMA gives 8x K-depth
// per instruction vs V_WMMA_F32_16X16X4_F32 while keeping f32 accumulation.
// Staging is coalesced + branchless (clamped addresses, float4 / packed-bf16
// LDS stores on the unit-stride paths).
// ============================================================================

typedef __attribute__((ext_vector_type(16))) __bf16 v16bf;
typedef __attribute__((ext_vector_type(8)))  float  v8f;

union FragAB { v16bf v; unsigned short u[16]; };
union FragC  { v8f  v; float f[8]; };

__device__ __forceinline__ unsigned short f2bf(float f) {
  unsigned int x = __float_as_uint(f);
  x += 0x7fffu + ((x >> 16) & 1u);          // round-to-nearest-even
  return (unsigned short)(x >> 16);
}

enum { GF_BIAS = 1, GF_RELU = 2, GF_ACC = 4 };

// ---------------------------------------------------------------------------
// Generic strided batched GEMM: C[b,i,j] = sum_k A[b,i,k]*B[b,k,j]  (+bias,
// +ReLU, +=C).  Strides in elements; aR/aC allow transposed A operands.
// Block = 128 threads = 4 wave32; 64x64 C tile; each wave owns 32x32 via 2x2
// v_wmma_f32_16x16x32_bf16 frags; K stepped by 32 through LDS (bf16 bits).
// ---------------------------------------------------------------------------
__global__ __launch_bounds__(128)
void k_wmma_gemm(const float* __restrict__ A, const float* __restrict__ Bm,
                 const float* __restrict__ bias, float* __restrict__ C,
                 int M, int N, int K,
                 long aB, long aR, long aC,
                 long bB, long bR, long bC,
                 long cB, int flags)
{
  __shared__ unsigned short As[64][34];     // [m][k] bf16 bits (+pad)
  __shared__ unsigned short Bs[64][34];     // [n][k] bf16 bits (transposed)
  const int b    = blockIdx.z;
  const int i0   = blockIdx.y * 64;
  const int j0   = blockIdx.x * 64;
  const int tid  = threadIdx.x;
  const int lane = tid & 31;
  const int wave = tid >> 5;
  const int wm   = (wave >> 1) * 32;
  const int wn   = (wave & 1) * 32;
  const float* Ab = A  + (long)b * aB;
  const float* Bb = Bm + (long)b * bB;
  float*       Cb = C  + (long)b * cB;

  const int coll  = lane & 15;              // C col / A row-in-frag / B col
  const int rbase = (lane < 16) ? 0 : 8;    // C/D VGPR row layout
  const int kbaseA = (lane < 16) ? 0 : 8;   // 16-bit A-matrix 16x32 layout
  const int khalfB = (lane < 16) ? 0 : 16;  // B-matrix K half

  FragC acc[2][2];
#pragma unroll
  for (int fm = 0; fm < 2; ++fm)
#pragma unroll
    for (int fn = 0; fn < 2; ++fn)
#pragma unroll
      for (int r = 0; r < 8; ++r) {
        float v = 0.0f;
        if (flags & GF_ACC) {
          int gr = i0 + wm + fm*16 + rbase + r;
          int gc = j0 + wn + fn*16 + coll;
          if (gr < M && gc < N) v = Cb[(long)gr * N + gc];
        }
        acc[fm][fn].f[r] = v;
      }

  for (int k0 = 0; k0 < K; k0 += 32) {
    // ---- stage A tile: As[r][c] = A[i0+r, k0+c] ----
    if (aC == 1) {
      // k is unit-stride in memory: float4 loads + packed bf16x2 LDS stores
      const bool full = (i0 + 64 <= M) && (k0 + 32 <= K) && ((aR & 3) == 0);
      for (int t = tid; t < 512; t += 128) {
        int r = t >> 3, c4 = (t & 7) << 2;
        int gi = i0 + r;
        long gk = k0 + c4;
        float4 v4;
        if (full) {
          v4 = *(const float4*)(Ab + (long)gi * aR + gk);
        } else {
          float tmp[4];
#pragma unroll
          for (int u = 0; u < 4; ++u) {
            bool ok = (gi < M) && (gk + u < K);
            long ai = ok ? ((long)gi * aR + gk + u) : 0;   // clamped, branchless
            float v = Ab[ai];
            tmp[u] = ok ? v : 0.0f;
          }
          v4.x = tmp[0]; v4.y = tmp[1]; v4.z = tmp[2]; v4.w = tmp[3];
        }
        unsigned int p0 = (unsigned)f2bf(v4.x) | ((unsigned)f2bf(v4.y) << 16);
        unsigned int p1 = (unsigned)f2bf(v4.z) | ((unsigned)f2bf(v4.w) << 16);
        unsigned int* d = (unsigned int*)&As[r][c4];       // 4B aligned (68r+8*(c4/4))
        d[0] = p0; d[1] = p1;
      }
    } else {
      // transposed A (aR==1): consecutive threads walk contiguous i (coalesced)
      for (int t = tid; t < 2048; t += 128) {
        int c = t >> 6, r = t & 63;
        int gi = i0 + r, gk = k0 + c;
        bool ok = (gi < M) && (gk < K);
        long ai = ok ? ((long)gi * aR + (long)gk * aC) : 0;
        float v = Ab[ai];
        As[r][c] = f2bf(ok ? v : 0.0f);
      }
    }
    // ---- stage B tile (transposed in LDS): Bs[n][c] = B[k0+c, j0+n] ----
    if (bC == 1) {
      const bool full = (j0 + 64 <= N) && (k0 + 32 <= K) && ((bR & 3) == 0);
      for (int t = tid; t < 512; t += 128) {
        int c = t >> 4, n4 = (t & 15) << 2;
        int gk = k0 + c, gj = j0 + n4;
        float4 v4;
        if (full) {
          v4 = *(const float4*)(Bb + (long)gk * bR + gj);
        } else {
          float tmp[4];
#pragma unroll
          for (int u = 0; u < 4; ++u) {
            bool ok = (gk < K) && (gj + u < N);
            long bi = ok ? ((long)gk * bR + gj + u) : 0;   // clamped, branchless
            float v = Bb[bi];
            tmp[u] = ok ? v : 0.0f;
          }
          v4.x = tmp[0]; v4.y = tmp[1]; v4.z = tmp[2]; v4.w = tmp[3];
        }
        Bs[n4 + 0][c] = f2bf(v4.x);
        Bs[n4 + 1][c] = f2bf(v4.y);
        Bs[n4 + 2][c] = f2bf(v4.z);
        Bs[n4 + 3][c] = f2bf(v4.w);
      }
    } else {
      for (int t = tid; t < 2048; t += 128) {
        int nI = t >> 5, c = t & 31;
        int gj = j0 + nI, gk = k0 + c;
        bool ok = (gj < N) && (gk < K);
        long bi = ok ? ((long)gk * bR + (long)gj * bC) : 0;
        float v = Bb[bi];
        Bs[nI][c] = f2bf(ok ? v : 0.0f);
      }
    }
    __syncthreads();

    FragAB af[2], bf2[2];
#pragma unroll
    for (int fm = 0; fm < 2; ++fm) {
      int row = wm + fm*16 + coll;
#pragma unroll
      for (int e = 0; e < 16; ++e) {
        int kk = khalfB + e;
        af[fm].u[e] = As[row][kk];
      }
    }
#pragma unroll
    for (int fn = 0; fn < 2; ++fn) {
      int col = wn + fn*16 + coll;
#pragma unroll
      for (int e = 0; e < 16; ++e)
        bf2[fn].u[e] = Bs[col][khalfB + e];
    }
#pragma unroll
    for (int fm = 0; fm < 2; ++fm)
#pragma unroll
      for (int fn = 0; fn < 2; ++fn)
        acc[fm][fn].v = __builtin_amdgcn_wmma_f32_16x16x32_bf16(
            false, af[fm].v, false, bf2[fn].v,
            (short)0, acc[fm][fn].v, false, false);
    __syncthreads();
  }

#pragma unroll
  for (int fm = 0; fm < 2; ++fm)
#pragma unroll
    for (int fn = 0; fn < 2; ++fn)
#pragma unroll
      for (int r = 0; r < 8; ++r) {
        int gr = i0 + wm + fm*16 + rbase + r;
        int gc = j0 + wn + fn*16 + coll;
        if (gr < M && gc < N) {
          float v = acc[fm][fn].f[r];
          if (flags & GF_BIAS) v += bias[gc];
          if (flags & GF_RELU) v = fmaxf(v, 0.0f);
          Cb[(long)gr * N + gc] = v;
        }
      }
}

// ---------------------------------------------------------------------------
// Small utility / graph kernels
// ---------------------------------------------------------------------------
__global__ void k_zero(float* p, long n) {
  long i = (long)blockIdx.x * 256 + threadIdx.x;
  if (i < n) p[i] = 0.0f;
}

__global__ void k_count_deg(const int* __restrict__ dst, float* deg, int E) {
  int e = blockIdx.x * 256 + threadIdx.x;
  if (e < E) atomicAdd(&deg[dst[e]], 1.0f);
}

// edge MLP: h = relu(attr@W1+b1)@W2+b2 ; acc[dst] += h   (weights in LDS)
__global__ void k_edge_mlp(const float* __restrict__ attr, const int* __restrict__ dst,
                           const float* __restrict__ w1, const float* __restrict__ b1,
                           const float* __restrict__ w2, const float* __restrict__ b2,
                           float* acc, int E)
{
  __shared__ float sw1[16*25], sb1[25], sw2[25*64], sb2[64];
  int tid = threadIdx.x;
  for (int i = tid; i < 16*25; i += 256) sw1[i] = w1[i];
  for (int i = tid; i < 25;    i += 256) sb1[i] = b1[i];
  for (int i = tid; i < 25*64; i += 256) sw2[i] = w2[i];
  for (int i = tid; i < 64;    i += 256) sb2[i] = b2[i];
  __syncthreads();
  int e = blockIdx.x * 256 + tid;
  if (e >= E) return;
  float av[16];
#pragma unroll
  for (int f = 0; f < 16; ++f) av[f] = attr[(long)e*16 + f];
  float hid[25];
  for (int u = 0; u < 25; ++u) {
    float s = sb1[u];
#pragma unroll
    for (int f = 0; f < 16; ++f) s += av[f] * sw1[f*25 + u];
    hid[u] = fmaxf(s, 0.0f);
  }
  float* row = acc + (long)dst[e] * 64;
  for (int h = 0; h < 64; ++h) {
    float s = sb2[h];
    for (int u = 0; u < 25; ++u) s += hid[u] * sw2[u*64 + h];
    atomicAdd(&row[h], s);
  }
}

__global__ void k_node_elu(const float* __restrict__ acc, const float* __restrict__ deg,
                           const float* __restrict__ root, const float* __restrict__ bias,
                           float* x, int Ntot)
{
  int nId = blockIdx.x, h = threadIdx.x;
  float v = acc[(long)nId*64 + h] / fmaxf(deg[nId], 1.0f) + root[h] + bias[h];
  x[(long)nId*64 + h] = v > 0.0f ? v : (expf(v) - 1.0f);   // ELU
}

__global__ void k_edge_agg(const float* __restrict__ x, const int* __restrict__ src,
                           const int* __restrict__ dst, float* acc, int E)
{
  int e = blockIdx.x, h = threadIdx.x;
  if (e < E) atomicAdd(&acc[(long)dst[e]*64 + h], x[(long)src[e]*64 + h]);
}

__global__ void k_scale_rows(float* a, const float* __restrict__ deg, int rows) {
  int r = blockIdx.x, h = threadIdx.x;
  if (r < rows) a[(long)r*64 + h] /= fmaxf(deg[r], 1.0f);
}

__global__ void k_mean_store(const float* __restrict__ x, float* z, int n, int layer) {
  int b = blockIdx.x, h = threadIdx.x;
  float s = 0.0f;
  for (int nd = 0; nd < n; ++nd) s += x[((long)b*n + nd)*64 + h];
  z[(long)b*640 + layer*64 + h] = s / (float)n;
}

__global__ void k_build_adj(const int* __restrict__ src, const int* __restrict__ dst,
                            float* A, int E)
{
  int e = blockIdx.x * 256 + threadIdx.x;
  if (e >= E) return;
  int d = dst[e], s = src[e];
  int g = d >> 9;                 // / 512
  int tl = d & 511, sl = s & 511;
  A[((long)g*512 + sl)*512 + tl] = 1.0f;
}

// A = where(eye & ~M, 1, A) * Mf(with diag); Mf |= eye.  In: A_io (A and M=A>0)
__global__ void k_prep_pool(float* A_io, float* A_out, int n, long total) {
  long e = (long)blockIdx.x * 256 + threadIdx.x;
  if (e >= total) return;
  long nn = (long)n * n;
  long rem = e % nn;
  int s = (int)(rem / n), t = (int)(rem % n);
  float a = A_io[e];
  bool mold = a > 0.0f, diag = (s == t);
  bool m = mold || diag;
  A_out[e] = m ? ((diag && !mold) ? 1.0f : a) : 0.0f;
  A_io[e]  = m ? 1.0f : 0.0f;     // becomes Mf
}

// xm[b,t,h] = max over s with Mf[b,s,t]>0 of x[b,s,h]  (else -1e30)
__global__ void k_masked_max(const float* __restrict__ x, const float* __restrict__ Mf,
                             float* xm, int n)
{
  int bt = blockIdx.x, h = threadIdx.x;
  int b = bt / n, t = bt - b*n;
  const float* Mb = Mf + (long)b*n*n + t;
  const float* xb = x  + (long)b*n*64;
  float v = -1e30f;
  for (int s = 0; s < n; ++s)
    if (Mb[(long)s*n] > 0.0f) v = fmaxf(v, xb[(long)s*64 + h]);
  xm[(long)bt*64 + h] = v;
}

__global__ void k_att(const float* __restrict__ x, const float* __restrict__ xq,
                      const float* __restrict__ attw, float* a_s, float* a_t, int total)
{
  int e = blockIdx.x * 128 + threadIdx.x;
  if (e >= total) return;
  float st = 0.0f, ss = 0.0f;
  const float* q = xq + (long)e*64;
  const float* xr = x + (long)e*64;
  for (int h = 0; h < 64; ++h) { st += q[h]*attw[h]; ss += xr[h]*attw[64+h]; }
  a_t[e] = st; a_s[e] = ss;
}

// column softmax over s: S[b,s,t] = softmax_s(mask? lrelu(a_s+a_t+b) : -1e30)*Mf
__global__ void k_softmax_col(const float* __restrict__ a_s, const float* __restrict__ a_t,
                              const float* __restrict__ attb, int j,
                              const float* __restrict__ Mf, float* S, int n)
{
  __shared__ float red[256];
  int bt = blockIdx.x, tid = threadIdx.x;
  int b = bt / n, t = bt - b*n;
  float ab = attb[j];
  float at = a_t[(long)b*n + t];
  const float* Mb = Mf + (long)b*n*n + t;
  const float* asb = a_s + (long)b*n;
  float mx = -1e30f;
  for (int s = tid; s < n; s += 256) {
    if (Mb[(long)s*n] > 0.0f) {
      float zv = asb[s] + at + ab;
      float lr = zv > 0.0f ? zv : 0.2f*zv;
      mx = fmaxf(mx, lr);
    }
  }
  red[tid] = mx; __syncthreads();
  for (int o = 128; o > 0; o >>= 1) { if (tid < o) red[tid] = fmaxf(red[tid], red[tid+o]); __syncthreads(); }
  mx = red[0]; __syncthreads();
  float sum = 0.0f;
  for (int s = tid; s < n; s += 256) {
    if (Mb[(long)s*n] > 0.0f) {
      float zv = asb[s] + at + ab;
      float lr = zv > 0.0f ? zv : 0.2f*zv;
      sum += expf(lr - mx);
    }
  }
  red[tid] = sum; __syncthreads();
  for (int o = 128; o > 0; o >>= 1) { if (tid < o) red[tid] += red[tid+o]; __syncthreads(); }
  float inv = 1.0f / red[0];
  float* Sb = S + (long)b*n*n + t;
  for (int s = tid; s < n; s += 256) {
    float out = 0.0f;
    if (Mb[(long)s*n] > 0.0f) {
      float zv = asb[s] + at + ab;
      float lr = zv > 0.0f ? zv : 0.2f*zv;
      out = expf(lr - mx) * inv;
    }
    Sb[(long)s*n] = out;
  }
}

__global__ void k_a123(const float* __restrict__ xc,
                       const float* __restrict__ w1, const float* __restrict__ b1v,
                       const float* __restrict__ w2, const float* __restrict__ w3,
                       const float* __restrict__ b3v, int j,
                       float* a1, float* a2, float* a3, int total)
{
  int e = blockIdx.x * 128 + threadIdx.x;
  if (e >= total) return;
  const float* r = xc + (long)e*64;
  float s1 = 0, s2 = 0, s3 = 0;
  for (int h = 0; h < 64; ++h) { float v = r[h]; s1 += v*w1[h]; s2 += v*w2[h]; s3 += v*w3[h]; }
  a1[e] = s1 + b1v[j]; a2[e] = s2; a3[e] = s3 + b3v[j];
}

__global__ void k_colsum_pos(const float* __restrict__ A, float* degt, int n, int total) {
  int e = blockIdx.x * 256 + threadIdx.x;
  if (e >= total) return;
  int b = e / n, t = e - b*n;
  const float* Ab = A + (long)b*n*n + t;
  float s = 0.0f;
  for (int ss = 0; ss < n; ++ss) s += (Ab[(long)ss*n] > 0.0f) ? 1.0f : 0.0f;
  degt[e] = s;
}

__global__ void k_fitness(const float* __restrict__ a1, const float* __restrict__ a2,
                          const float* __restrict__ a3, const float* __restrict__ degt,
                          const float* __restrict__ Mf, float* fit, int n, int total)
{
  int e = blockIdx.x * 256 + threadIdx.x;
  if (e >= total) return;
  int b = e / n, t = e - b*n;
  const float* Mb = Mf + (long)b*n*n + t;
  const float* a2b = a2 + (long)b*n;
  float s = 0.0f;
  for (int ss = 0; ss < n; ++ss) s += Mb[(long)ss*n] * a2b[ss];
  float v = a3[e] + degt[e]*a1[e] - s;
  fit[e] = 1.0f / (1.0f + expf(-v));
}

// stable top-k via bitonic sort on (value desc, index asc); n<=512, per-graph
__global__ void k_topk(const float* __restrict__ fit, float* vals, int* idx, int n, int k) {
  __shared__ float sv[512];
  __shared__ int   si[512];
  int b = blockIdx.x, tid = threadIdx.x;
  for (int i = tid; i < 512; i += 256) {
    sv[i] = (i < n) ? fit[(long)b*n + i] : -3.0e38f;
    si[i] = i;
  }
  __syncthreads();
  for (int size = 2; size <= 512; size <<= 1) {
    for (int stride = size >> 1; stride > 0; stride >>= 1) {
      for (int i = tid; i < 512; i += 256) {
        int p = i ^ stride;
        if (p > i) {
          bool gt_p = (sv[p] > sv[i]) || (sv[p] == sv[i] && si[p] < si[i]);
          bool desc = ((i & size) == 0);
          if (desc == gt_p) {
            float tv = sv[i]; sv[i] = sv[p]; sv[p] = tv;
            int   ti = si[i]; si[i] = si[p]; si[p] = ti;
          }
        }
      }
      __syncthreads();
    }
  }
  for (int i = tid; i < k; i += 256) {
    vals[(long)b*512 + i] = sv[i];
    idx [(long)b*512 + i] = si[i];
  }
}

__global__ void k_gather_x(const float* __restrict__ xc, const int* __restrict__ idx,
                           const float* __restrict__ vals, float* xnew, int n, int k)
{
  int bi = blockIdx.x, h = threadIdx.x;
  int b = bi / k, i = bi - b*k;
  int t = idx[(long)b*512 + i];
  float v = vals[(long)b*512 + i];
  xnew[((long)b*k + i)*64 + h] = xc[((long)b*n + t)*64 + h] * v;
}

__global__ void k_gather_S(const float* __restrict__ S, const int* __restrict__ idx,
                           float* Ssel, int n, int k, long total)
{
  long e = (long)blockIdx.x * 256 + threadIdx.x;
  if (e >= total) return;
  long nk = (long)n * k;
  int b = (int)(e / nk);
  long rem = e - (long)b*nk;
  int s = (int)(rem / k), i = (int)(rem - (long)s*k);
  Ssel[e] = S[(long)b*n*n + (long)s*n + idx[(long)b*512 + i]];
}

__global__ void k_zero_diag(float* A, int k, int total) {
  int e = blockIdx.x * 256 + threadIdx.x;
  if (e >= total) return;
  int b = e / k, i = e - b*k;
  A[(long)b*k*k + (long)i*k + i] = 0.0f;
}

__global__ void k_head1(const float* __restrict__ z, const float* __restrict__ w,
                        const float* __restrict__ bias, float* zz)
{
  int t = blockIdx.x * 128 + threadIdx.x;
  if (t >= 8*64) return;
  int b = t >> 6, o = t & 63;
  float s = bias[o];
  const float* zr = z + (long)b*640;
  for (int i = 0; i < 640; ++i) s += zr[i] * w[(long)i*64 + o];
  zz[t] = fmaxf(s, 0.0f);
}

__global__ void k_head2(const float* __restrict__ zz, const float* __restrict__ w,
                        const float* __restrict__ bias, float* out)
{
  __shared__ float lg[10];
  __shared__ float smx, slse;
  int b = blockIdx.x, tid = threadIdx.x;
  if (tid < 10) {
    float s = bias[tid];
    for (int i = 0; i < 64; ++i) s += zz[(long)b*64 + i] * w[(long)i*10 + tid];
    lg[tid] = s;
  }
  __syncthreads();
  if (tid == 0) {
    float mx = lg[0];
    for (int o = 1; o < 10; ++o) mx = fmaxf(mx, lg[o]);
    float sm = 0.0f;
    for (int o = 0; o < 10; ++o) sm += expf(lg[o] - mx);
    smx = mx; slse = logf(sm);
  }
  __syncthreads();
  if (tid < 10) out[(long)b*10 + tid] = lg[tid] - smx - slse;
}

// ---------------------------------------------------------------------------
// Host helpers
// ---------------------------------------------------------------------------
static inline void gemm(hipStream_t st, int batch,
                        const float* A, const float* Bm, const float* bias, float* C,
                        int M, int N, int K,
                        long aB, long aR, long aC,
                        long bB, long bR, long bC,
                        long cB, int flags)
{
  dim3 grid((N + 63) / 64, (M + 63) / 64, batch);
  k_wmma_gemm<<<grid, 128, 0, st>>>(A, Bm, bias, C, M, N, K,
                                    aB, aR, aC, bB, bR, bC, cB, flags);
}

extern "C" void kernel_launch(void* const* d_in, const int* in_sizes, int n_in,
                              void* d_out, int out_size, void* d_ws, size_t ws_size,
                              hipStream_t stream)
{
  const int B = 8, N0 = 512, H = 64, Ntot = B * N0;
  const int KS[4] = {461, 415, 374, 337};
  const int E = in_sizes[1];                       // 65536

  const float* edge_attr = (const float*)d_in[0];
  const int*   edge_src  = (const int*)  d_in[1];
  const int*   edge_dst  = (const int*)  d_in[2];
  const float* mlp_w1    = (const float*)d_in[3];
  const float* mlp_b1    = (const float*)d_in[4];
  const float* mlp_w2    = (const float*)d_in[5];
  const float* mlp_b2    = (const float*)d_in[6];
  const float* nn_root   = (const float*)d_in[7];
  const float* nn_bias   = (const float*)d_in[8];
  const float* c1_wrel   = (const float*)d_in[9];
  const float* c1_brel   = (const float*)d_in[10];
  const float* c1_wroot  = (const float*)d_in[11];
  const float* cv_wrel   = (const float*)d_in[12]; // (9,64,64)
  const float* cv_brel   = (const float*)d_in[13]; // (9,64)
  const float* cv_wroot  = (const float*)d_in[14];
  const float* pl_w      = (const float*)d_in[15]; // (4,64,64)
  const float* pl_b      = (const float*)d_in[16]; // (4,64)
  const float* pa_w      = (const float*)d_in[17]; // (4,128)
  const float* pa_b      = (const float*)d_in[18]; // (4,)
  const float* le1_w     = (const float*)d_in[19];
  const float* le1_b     = (const float*)d_in[20];
  const float* le2_w     = (const float*)d_in[21];
  const float* le3_w     = (const float*)d_in[22];
  const float* le3_b     = (const float*)d_in[23];
  const float* lin1_w    = (const float*)d_in[24];
  const float* lin1_b    = (const float*)d_in[25];
  const float* lin2_w    = (const float*)d_in[26];
  const float* lin2_b    = (const float*)d_in[27];

  // ---- workspace arena (~40 MB of floats) ----
  float* base = (float*)d_ws;
  size_t off = 0;
  auto take = [&](size_t nf) { float* p = base + off; off += nf; return p; };
  const long BNN = (long)B * 512 * 512;
  float* W0  = take(BNN);        // adjacency A / Mf
  float* W1  = take(BNN);        // masked A for products
  float* W2  = take(BNN);        // S, then tmp = S_selT A
  float* W3  = take(BNN);        // S_sel
  float* xA  = take((size_t)Ntot * H);
  float* xB  = take((size_t)Ntot * H);
  float* acc = take((size_t)Ntot * H);
  float* xm  = take((size_t)Ntot * H);
  float* xq  = take((size_t)Ntot * H);
  float* xc  = take((size_t)Ntot * H);
  float* degN = take(Ntot);
  float* a_s  = take((size_t)B * 512);
  float* a_t  = take((size_t)B * 512);
  float* a1   = take((size_t)B * 512);
  float* a2   = take((size_t)B * 512);
  float* a3   = take((size_t)B * 512);
  float* degt = take((size_t)B * 512);
  float* fit  = take((size_t)B * 512);
  float* vals = take((size_t)B * 512);
  int*   idx  = (int*)take((size_t)B * 512);
  float* zbuf = take((size_t)B * 640);
  float* zz   = take((size_t)B * 64);

  float* xcur = xA;
  float* xalt = xB;

  // ---------------- edge MLP -> node features ----------------
  k_zero<<<(Ntot + 255) / 256, 256, 0, stream>>>(degN, Ntot);
  k_zero<<<((long)Ntot * H + 255) / 256, 256, 0, stream>>>(acc, (long)Ntot * H);
  k_count_deg<<<(E + 255) / 256, 256, 0, stream>>>(edge_dst, degN, E);
  k_edge_mlp<<<(E + 255) / 256, 256, 0, stream>>>(edge_attr, edge_dst,
                                                  mlp_w1, mlp_b1, mlp_w2, mlp_b2, acc, E);
  k_node_elu<<<Ntot, 64, 0, stream>>>(acc, degN, nn_root, nn_bias, xcur, Ntot);

  // ---------------- sparse conv 1 (c1_*) ----------------
  k_zero<<<((long)Ntot * H + 255) / 256, 256, 0, stream>>>(acc, (long)Ntot * H);
  k_edge_agg<<<E, 64, 0, stream>>>(xcur, edge_src, edge_dst, acc, E);
  k_scale_rows<<<Ntot, 64, 0, stream>>>(acc, degN, Ntot);
  gemm(stream, 1, acc, c1_wrel, c1_brel, xalt, Ntot, 64, 64, 0, 64, 1, 0, 64, 1, 0, GF_BIAS);
  gemm(stream, 1, xcur, c1_wroot, nullptr, xalt, Ntot, 64, 64, 0, 64, 1, 0, 64, 1, 0, GF_ACC | GF_RELU);
  { float* t = xcur; xcur = xalt; xalt = t; }
  k_mean_store<<<B, 64, 0, stream>>>(xcur, zbuf, N0, 0);

  // ---------------- sparse conv 2 (convs[0]) ----------------
  k_zero<<<((long)Ntot * H + 255) / 256, 256, 0, stream>>>(acc, (long)Ntot * H);
  k_edge_agg<<<E, 64, 0, stream>>>(xcur, edge_src, edge_dst, acc, E);
  k_scale_rows<<<Ntot, 64, 0, stream>>>(acc, degN, Ntot);
  gemm(stream, 1, acc, cv_wrel, cv_brel, xalt, Ntot, 64, 64, 0, 64, 1, 0, 64, 1, 0, GF_BIAS);
  gemm(stream, 1, xcur, cv_wroot, nullptr, xalt, Ntot, 64, 64, 0, 64, 1, 0, 64, 1, 0, GF_ACC | GF_RELU);
  { float* t = xcur; xcur = xalt; xalt = t; }
  k_mean_store<<<B, 64, 0, stream>>>(xcur, zbuf, N0, 1);

  // ---------------- dense adjacency ----------------
  k_zero<<<(BNN + 255) / 256, 256, 0, stream>>>(W0, BNN);
  k_build_adj<<<(E + 255) / 256, 256, 0, stream>>>(edge_src, edge_dst, W0, E);

  int n = N0;

  // ---------------- ASAP pool (lambda over current n, j, k) ----------------
  auto do_pool = [&](int j, int k) {
    long tot = (long)B * n * n;
    k_prep_pool<<<(tot + 255) / 256, 256, 0, stream>>>(W0, W1, n, tot);   // W0=Mf, W1=A
    k_masked_max<<<B * n, 64, 0, stream>>>(xcur, W0, xm, n);
    gemm(stream, B, xm, pl_w + (long)j * 4096, pl_b + (long)j * 64, xq,
         n, 64, 64, (long)n * 64, 64, 1, 0, 64, 1, (long)n * 64, GF_BIAS);
    k_att<<<(B * n + 127) / 128, 128, 0, stream>>>(xcur, xq, pa_w + (long)j * 128,
                                                   a_s, a_t, B * n);
    k_softmax_col<<<B * n, 256, 0, stream>>>(a_s, a_t, pa_b, j, W0, W2, n);
    // xc = S^T x
    gemm(stream, B, W2, xcur, nullptr, xc, n, 64, n,
         (long)n * n, 1, n, (long)n * 64, 64, 1, (long)n * 64, 0);
    k_a123<<<(B * n + 127) / 128, 128, 0, stream>>>(xc,
         le1_w + (long)j * 64, le1_b, le2_w + (long)j * 64,
         le3_w + (long)j * 64, le3_b, j, a1, a2, a3, B * n);
    k_colsum_pos<<<(B * n + 255) / 256, 256, 0, stream>>>(W0, degt, n, B * n);
    k_fitness<<<(B * n + 255) / 256, 256, 0, stream>>>(a1, a2, a3, degt, W0, fit, n, B * n);
    k_topk<<<B, 256, 0, stream>>>(fit, vals, idx, n, k);
    k_gather_x<<<B * k, 64, 0, stream>>>(xc, idx, vals, xalt, n, k);
    long totS = (long)B * n * k;
    k_gather_S<<<(totS + 255) / 256, 256, 0, stream>>>(W2, idx, W3, n, k, totS);
    // tmp = S_sel^T A   (k x n)
    gemm(stream, B, W3, W1, nullptr, W2, k, n, n,
         (long)n * k, 1, k, (long)n * n, n, 1, (long)k * n, 0);
    // A_new = tmp @ S_sel   (k x k)
    gemm(stream, B, W2, W3, nullptr, W0, k, k, n,
         (long)k * n, n, 1, (long)n * k, k, 1, (long)k * k, 0);
    k_zero_diag<<<(B * k + 255) / 256, 256, 0, stream>>>(W0, k, B * k);
    { float* t = xcur; xcur = xalt; xalt = t; }
    n = k;
  };

  do_pool(0, KS[0]);

  // ---------------- dense conv + pool loop (i = 1..8) ----------------
  for (int i = 1; i <= 8; ++i) {
    const float* wrel  = cv_wrel  + (long)i * 4096;
    const float* brel  = cv_brel  + (long)i * 64;
    const float* wroot = cv_wroot + (long)i * 4096;
    k_colsum_pos<<<(B * n + 255) / 256, 256, 0, stream>>>(W0, degt, n, B * n);
    // agg = A^T x
    gemm(stream, B, W0, xcur, nullptr, acc, n, 64, n,
         (long)n * n, 1, n, (long)n * 64, 64, 1, (long)n * 64, 0);
    k_scale_rows<<<B * n, 64, 0, stream>>>(acc, degt, B * n);
    gemm(stream, B, acc, wrel, brel, xalt, n, 64, 64,
         (long)n * 64, 64, 1, 0, 64, 1, (long)n * 64, GF_BIAS);
    gemm(stream, B, xcur, wroot, nullptr, xalt, n, 64, 64,
         (long)n * 64, 64, 1, 0, 64, 1, (long)n * 64, GF_ACC | GF_RELU);
    { float* t = xcur; xcur = xalt; xalt = t; }
    k_mean_store<<<B, 64, 0, stream>>>(xcur, zbuf, n, i + 1);
    if ((i == 2 || i == 4 || i == 6)) do_pool(i / 2, KS[i / 2]);
  }

  // ---------------- head ----------------
  k_head1<<<(8 * 64 + 127) / 128, 128, 0, stream>>>(zbuf, lin1_w, lin1_b, zz);
  k_head2<<<B, 32, 0, stream>>>(zz, lin2_w, lin2_b, (float*)d_out);
  (void)n_in; (void)out_size; (void)ws_size; (void)in_sizes;
}
